// ArcMarginProduct_6253472383512
// MI455X (gfx1250) — matrix operand order repaired and checked by reading, not compiled, not run on hardware
//
#include <hip/hip_runtime.h>
#include <hip/hip_bf16.h>

// ---------------- problem constants ----------------
#define B_ROWS 1024
#define D_DIM  512
#define C_COLS 100000
#define SCALE_S 30.0f
#define COSM_C 0.95533648912560601964f   // cos(0.3)
#define SINM_C 0.29552020666133957511f   // sin(0.3)
#define TH_C  (-0.95533648912560601964f) // cos(pi-0.3)
#define MM_C   0.08865606199840187253f   // sin(pi-0.3)*0.3
#define EPS_N  1e-12f

// ---------------- tiling ----------------
#define BM 128           // rows of features per block
#define BN 128           // weight rows (classes) per chunk
#define BK 32            // K step (bf16 WMMA K)
#define KI (D_DIM / BK)  // 16 K iterations
#define LDT 40           // LDS row stride in bf16 elems (80B, 16B aligned)
#define CHUNKS_PER_BLK 8
#define NPART 98         // 98*8*128 = 100352 >= 100000

typedef __attribute__((ext_vector_type(16))) __bf16 v16bf;
typedef __attribute__((ext_vector_type(8)))  float  v8f;

union Frag16 { v16bf v; uint4 q[2]; };
union Acc8   { v8f  v; float f[8]; };

// monotonic float<->uint encoding so atomicMax(u32) == float max
__device__ __forceinline__ unsigned int fenc(float f) {
  unsigned int u = __float_as_uint(f);
  return (u & 0x80000000u) ? ~u : (u | 0x80000000u);
}
__device__ __forceinline__ float fdec(unsigned int u) {
  unsigned int v = (u & 0x80000000u) ? (u & 0x7FFFFFFFu) : ~u;
  return __uint_as_float(v);
}
#define ENC_NEG_INF 0x007FFFFFu  // fenc(-inf)

// native f32 -> bf16 (clang lowers fptrunc-to-bf16 on gfx1250)
__device__ __forceinline__ unsigned int pack_bf16x2(float a, float b) {
  union { __bf16 h[2]; unsigned int u; } r;
  r.h[0] = (__bf16)a;
  r.h[1] = (__bf16)b;
  return r.u;
}

__device__ __forceinline__ float arc_logit(float c, bool isTarget) {
  float lg = c;
  if (isTarget) {
    float sn  = sqrtf(fmaxf(1.0f - c * c, 0.0f));
    float phi = c * COSM_C - sn * SINM_C;
    lg = (c > TH_C) ? phi : (c - MM_C);
  }
  return lg * SCALE_S;
}

// ---- CDNA5 async global->LDS copy (GVS mode), tracked by ASYNCcnt ----
// vdst = 32-bit LDS byte address, vaddr = per-lane 32-bit byte offset,
// saddr = uniform 64-bit base (SGPR pair)
__device__ __forceinline__ void async_copy_b128(unsigned int ldsByteAddr,
                                                const void* base,
                                                unsigned int byteOff) {
  asm volatile("global_load_async_to_lds_b128 %0, %1, %2"
               :
               : "v"(ldsByteAddr), "v"(byteOff), "s"(base)
               : "memory");
}
__device__ __forceinline__ void wait_async0() {
  asm volatile("s_wait_asynccnt 0x0" ::: "memory");
}
__device__ __forceinline__ unsigned int lds_addr(const void* p) {
  return (unsigned int)(size_t)p;  // low 32 bits of generic addr = LDS offset
}

// ---------------- kernel 1: L2-normalize rows of [N,512] f32 -> bf16 ----------------
// one wave (32 lanes) per row; lane handles 16 consecutive elems
__global__ __launch_bounds__(256)
void rownorm_bf16_kernel(const float* __restrict__ src,
                         unsigned short* __restrict__ dst) {
  const int wid  = threadIdx.x >> 5;
  const int lane = threadIdx.x & 31;
  const int row  = blockIdx.x * 8 + wid;

  const float* s = src + (size_t)row * D_DIM + lane * 16;
  float v[16];
  const float4* s4 = (const float4*)s;
  float4 a0 = s4[0], a1 = s4[1], a2 = s4[2], a3 = s4[3];
  v[0]=a0.x; v[1]=a0.y; v[2]=a0.z; v[3]=a0.w;
  v[4]=a1.x; v[5]=a1.y; v[6]=a1.z; v[7]=a1.w;
  v[8]=a2.x; v[9]=a2.y; v[10]=a2.z; v[11]=a2.w;
  v[12]=a3.x; v[13]=a3.y; v[14]=a3.z; v[15]=a3.w;

  float ss = 0.0f;
  #pragma unroll
  for (int i = 0; i < 16; ++i) ss += v[i] * v[i];
  #pragma unroll
  for (int o = 16; o > 0; o >>= 1) ss += __shfl_xor(ss, o, 32);  // wave32

  const float inv = 1.0f / fmaxf(sqrtf(ss), EPS_N);

  unsigned int p[8];
  #pragma unroll
  for (int i = 0; i < 8; ++i)
    p[i] = pack_bf16x2(v[2 * i] * inv, v[2 * i + 1] * inv);

  uint4* d = (uint4*)(dst + (size_t)row * D_DIM + lane * 16);
  d[0] = make_uint4(p[0], p[1], p[2], p[3]);
  d[1] = make_uint4(p[4], p[5], p[6], p[7]);
}

// ---------------- kernel 2 (fallback path): weight row inverse norms ----------------
__global__ __launch_bounds__(256)
void w_norm_kernel(const float* __restrict__ w, float* __restrict__ winv) {
  const int wid  = threadIdx.x >> 5;
  const int lane = threadIdx.x & 31;
  const int row  = blockIdx.x * 8 + wid;
  if (row >= C_COLS) return;

  const float4* src = (const float4*)(w + (size_t)row * D_DIM);
  float ss = 0.0f;
  #pragma unroll
  for (int q = 0; q < 4; ++q) {
    float4 a = src[lane + q * 32];
    ss += a.x * a.x + a.y * a.y + a.z * a.z + a.w * a.w;
  }
  #pragma unroll
  for (int o = 16; o > 0; o >>= 1) ss += __shfl_xor(ss, o, 32);

  if (lane == 0) winv[row] = 1.0f / fmaxf(sqrtf(ss), EPS_N);
}

// ---------------- kernel 3: fused WMMA GEMM + margin + online LSE ----------------
// PRE=true : B sourced from pre-normalized bf16 weight (async DMA staging)
// PRE=false: B converted f32->bf16 in-loop with winv scaling (small-ws fallback)
template <bool PRE>
__global__ __launch_bounds__(256)
void arc_gemm_kernel(const unsigned short* __restrict__ featbf,
                     const unsigned short* __restrict__ wbf,
                     const float* __restrict__ weight,
                     const float* __restrict__ winv,
                     const int* __restrict__ targets,
                     float* __restrict__ tlogit,
                     float* __restrict__ partMax,
                     float* __restrict__ partSum) {
  __shared__ unsigned short Al[2][BM * LDT];   // double-buffered A
  __shared__ unsigned short Bl[2][BN * LDT];   // double-buffered B
  __shared__ float        runMaxF[BM];
  __shared__ float        runSum[BM];
  __shared__ unsigned int chunkMaxB[BM];
  __shared__ int          rowTarget[BM];

  const int tid   = threadIdx.x;
  const int lane  = tid & 31;
  const int wid   = tid >> 5;
  const int waveM = wid & 1;   // 2 row groups of 64
  const int waveN = wid >> 1;  // 4 col groups of 32
  const int ln16  = lane & 15;
  const int halfL = lane >> 4;
  const int row0  = blockIdx.y * BM;

  // staging maps (A and bf16-B identical): row = tid/2, 2x16B per thread
  const int arow     = tid >> 1;
  const int asegBase = (tid & 1) * 2;  // uint4 segs {0,1} or {2,3}
  const int brow  = tid >> 1;
  const int bhalf = tid & 1;           // 16-elem half of the 32-elem row

  // per-thread global byte offsets (k=0)
  const unsigned int gaOff =
      (unsigned int)(((row0 + arow) * D_DIM + asegBase * 8) * sizeof(unsigned short));

  if (tid < BM) {
    runMaxF[tid]  = -__builtin_inff();
    runSum[tid]   = 0.0f;
    rowTarget[tid] = targets[row0 + tid];
  }
  __syncthreads();

  for (int chunk = 0; chunk < CHUNKS_PER_BLK; ++chunk) {
    const int c0 = (blockIdx.x * CHUNKS_PER_BLK + chunk) * BN;
    if (c0 >= C_COLS) break;  // uniform per block

    const int  bcol   = c0 + brow;
    const bool bvalid = (bcol < C_COLS);
    const float bscale = (!PRE && bvalid) ? winv[bcol] : 0.0f;
    const unsigned int gbOff =
        (unsigned int)(((size_t)bcol * D_DIM + bhalf * 16) * sizeof(unsigned short));

    // warm L2 for the NEXT column chunk's weight rows (global_prefetch_b8)
    {
      const int pcol = c0 + BN + brow;
      if (pcol < C_COLS) {
        if (PRE) __builtin_prefetch(wbf + (size_t)pcol * D_DIM, 0, 1);
        else     __builtin_prefetch(weight + (size_t)pcol * D_DIM, 0, 1);
      }
    }

    Acc8 acc[4][2];
    #pragma unroll
    for (int m = 0; m < 4; ++m)
      #pragma unroll
      for (int n = 0; n < 2; ++n)
        #pragma unroll
        for (int i = 0; i < 8; ++i) acc[m][n].f[i] = 0.0f;

    // ---- prologue: stage k-chunk 0 into buffer 0 ----
    {
      const unsigned int la = lds_addr(&Al[0][arow * LDT + asegBase * 8]);
      async_copy_b128(la,      featbf, gaOff);
      async_copy_b128(la + 16, featbf, gaOff + 16);
      if (PRE) {
        if (bvalid) {
          const unsigned int lb = lds_addr(&Bl[0][brow * LDT + bhalf * 16]);
          async_copy_b128(lb,      wbf, gbOff);
          async_copy_b128(lb + 16, wbf, gbOff + 16);
        }
      } else {
        float4 f0 = {0,0,0,0}, f1 = {0,0,0,0}, f2 = {0,0,0,0}, f3 = {0,0,0,0};
        if (bvalid) {
          const float4* gB = (const float4*)(weight + (size_t)bcol * D_DIM + bhalf * 16);
          f0 = gB[0]; f1 = gB[1]; f2 = gB[2]; f3 = gB[3];
        }
        *(uint4*)&Bl[0][brow * LDT + bhalf * 16] =
            make_uint4(pack_bf16x2(f0.x * bscale, f0.y * bscale),
                       pack_bf16x2(f0.z * bscale, f0.w * bscale),
                       pack_bf16x2(f1.x * bscale, f1.y * bscale),
                       pack_bf16x2(f1.z * bscale, f1.w * bscale));
        *(uint4*)&Bl[0][brow * LDT + bhalf * 16 + 8] =
            make_uint4(pack_bf16x2(f2.x * bscale, f2.y * bscale),
                       pack_bf16x2(f2.z * bscale, f2.w * bscale),
                       pack_bf16x2(f3.x * bscale, f3.y * bscale),
                       pack_bf16x2(f3.z * bscale, f3.w * bscale));
      }
      wait_async0();
    }
    __syncthreads();

    // ---- pipelined K loop: async DMA staging overlaps WMMA, 1 barrier/iter ----
    for (int ki = 0; ki < KI; ++ki) {
      const int p = ki & 1;
      const int q = p ^ 1;
      const bool more = (ki + 1 < KI);

      // 1) kick off async staging of the next k-chunk into buffer q
      float4 nB0 = {0,0,0,0}, nB1 = {0,0,0,0},
             nB2 = {0,0,0,0}, nB3 = {0,0,0,0};      // fallback path regs
      if (more) {
        const unsigned int knB = (unsigned int)((ki + 1) * BK * sizeof(unsigned short));
        const unsigned int la  = lds_addr(&Al[q][arow * LDT + asegBase * 8]);
        async_copy_b128(la,      featbf, gaOff + knB);
        async_copy_b128(la + 16, featbf, gaOff + knB + 16);
        if (PRE) {
          if (bvalid) {
            const unsigned int lb = lds_addr(&Bl[q][brow * LDT + bhalf * 16]);
            async_copy_b128(lb,      wbf, gbOff + knB);
            async_copy_b128(lb + 16, wbf, gbOff + knB + 16);
          }
        } else if (bvalid) {
          const int kn = (ki + 1) * BK;
          const float4* gB =
              (const float4*)(weight + (size_t)bcol * D_DIM + kn + bhalf * 16);
          nB0 = gB[0]; nB1 = gB[1]; nB2 = gB[2]; nB3 = gB[3];
        }
      }

      // 2) fragment loads (ds_load_b128) + 8 WMMAs from buffer p
      Frag16 afr[4], bfr[2];
      #pragma unroll
      for (int m = 0; m < 4; ++m) {
        const unsigned short* base = &Al[p][(waveM * 64 + m * 16 + ln16) * LDT];
        afr[m].q[0] = *(const uint4*)(base + halfL * 8);
        afr[m].q[1] = *(const uint4*)(base + 16 + halfL * 8);
      }
      #pragma unroll
      for (int n = 0; n < 2; ++n) {
        const unsigned short* base =
            &Bl[p][(waveN * 32 + n * 16 + ln16) * LDT];
        bfr[n].q[0] = *(const uint4*)(base + halfL * 8);
        bfr[n].q[1] = *(const uint4*)(base + 16 + halfL * 8);
      }
      #pragma unroll
      for (int m = 0; m < 4; ++m)
        #pragma unroll
        for (int n = 0; n < 2; ++n)
          acc[m][n].v = __builtin_amdgcn_wmma_f32_16x16x32_bf16(
              false, afr[m].v, false, bfr[n].v,
              (short)0, acc[m][n].v, false, false);

      // 3) fallback path: convert + store next B chunk
      if (more && !PRE) {
        *(uint4*)&Bl[q][brow * LDT + bhalf * 16] =
            make_uint4(pack_bf16x2(nB0.x * bscale, nB0.y * bscale),
                       pack_bf16x2(nB0.z * bscale, nB0.w * bscale),
                       pack_bf16x2(nB1.x * bscale, nB1.y * bscale),
                       pack_bf16x2(nB1.z * bscale, nB1.w * bscale));
        *(uint4*)&Bl[q][brow * LDT + bhalf * 16 + 8] =
            make_uint4(pack_bf16x2(nB2.x * bscale, nB2.y * bscale),
                       pack_bf16x2(nB2.z * bscale, nB2.w * bscale),
                       pack_bf16x2(nB3.x * bscale, nB3.y * bscale),
                       pack_bf16x2(nB3.z * bscale, nB3.w * bscale));
      }

      // 4) publish staged buffer
      wait_async0();
      __syncthreads();
    }

    // ---- epilogue pass A: margin once, overwrite acc with logits, chunk max ----
    if (tid < BM) chunkMaxB[tid] = ENC_NEG_INF;
    __syncthreads();
    #pragma unroll
    for (int m = 0; m < 4; ++m) {
      #pragma unroll
      for (int v = 0; v < 8; ++v) {
        const int rL = waveM * 64 + m * 16 + halfL * 8 + v;
        const int rT = rowTarget[rL];
        float mx = -__builtin_inff();
        #pragma unroll
        for (int n = 0; n < 2; ++n) {
          const int col = c0 + waveN * 32 + n * 16 + ln16;
          float lg = -__builtin_inff();
          if (col < C_COLS) {
            const bool isT = (col == rT);
            lg = arc_logit(acc[m][n].f[v], isT);
            if (isT) tlogit[row0 + rL] = lg;  // exactly one writer per row
          }
          acc[m][n].f[v] = lg;               // cache logit for pass B
          mx = fmaxf(mx, lg);
        }
        atomicMax(&chunkMaxB[rL], fenc(mx)); // ds_max_u32
      }
    }
    __syncthreads();
    // merge chunk max into running (max, sum) — one thread per row
    if (tid < BM) {
      const float cm = fdec(chunkMaxB[tid]);
      const float nm = fmaxf(runMaxF[tid], cm);
      runSum[tid] *= __expf(runMaxF[tid] - nm);  // exp(-inf)=0 first time
      runMaxF[tid] = nm;
    }
    __syncthreads();
    // ---- epilogue pass B: exp-sum (ds_add_f32), logits already cached ----
    #pragma unroll
    for (int m = 0; m < 4; ++m) {
      #pragma unroll
      for (int v = 0; v < 8; ++v) {
        const int rL = waveM * 64 + m * 16 + halfL * 8 + v;
        const float rm = runMaxF[rL];
        const float e = __expf(acc[m][0].f[v] - rm) + __expf(acc[m][1].f[v] - rm);
        atomicAdd(&runSum[rL], e);
      }
    }
    __syncthreads();
  }

  if (tid < BM) {
    const size_t idx = (size_t)(row0 + tid) * NPART + blockIdx.x;
    partMax[idx] = runMaxF[tid];
    partSum[idx] = runSum[tid];
  }
}

// ---------------- kernel 4: final reduction ----------------
__global__ __launch_bounds__(256)
void finalize_kernel(const float* __restrict__ partMax,
                     const float* __restrict__ partSum,
                     const float* __restrict__ tlogit,
                     float* __restrict__ out) {
  __shared__ float red[256];
  float accum = 0.0f;
  for (int r = threadIdx.x; r < B_ROWS; r += 256) {
    const float* pm = partMax + (size_t)r * NPART;
    const float* ps = partSum + (size_t)r * NPART;
    float M = -__builtin_inff();
    for (int p = 0; p < NPART; ++p) M = fmaxf(M, pm[p]);
    float Ssum = 0.0f;
    for (int p = 0; p < NPART; ++p) Ssum += ps[p] * __expf(pm[p] - M);
    accum += (M + logf(Ssum)) - tlogit[r];
  }
  red[threadIdx.x] = accum;
  __syncthreads();
  for (int s = 128; s > 0; s >>= 1) {
    if (threadIdx.x < s) red[threadIdx.x] += red[threadIdx.x + s];
    __syncthreads();
  }
  if (threadIdx.x == 0) out[0] = red[0] / (float)B_ROWS;
}

// ---------------- launch ----------------
extern "C" void kernel_launch(void* const* d_in, const int* in_sizes, int n_in,
                              void* d_out, int out_size, void* d_ws, size_t ws_size,
                              hipStream_t stream) {
  const float* features = (const float*)d_in[0];  // [1024,512] f32
  const float* weight   = (const float*)d_in[1];  // [100000,512] f32
  const int*   targets  = (const int*)d_in[2];    // [1024] int
  float* out = (float*)d_out;

  char* ws = (char*)d_ws;
  size_t off = 0;
  auto alignup = [](size_t x) { return (x + 255) & ~(size_t)255; };

  // small buffers first so the fallback path fits in a small workspace
  unsigned short* featbf = (unsigned short*)(ws + off);
  off = alignup(off + (size_t)B_ROWS * D_DIM * sizeof(unsigned short));
  float* tlog = (float*)(ws + off);
  off = alignup(off + (size_t)B_ROWS * sizeof(float));
  float* pMax = (float*)(ws + off);
  off = alignup(off + (size_t)B_ROWS * NPART * sizeof(float));
  float* pSum = (float*)(ws + off);
  off = alignup(off + (size_t)B_ROWS * NPART * sizeof(float));
  float* winv = (float*)(ws + off);
  off = alignup(off + (size_t)C_COLS * sizeof(float));
  const size_t need_fallback = off;
  unsigned short* wbf = (unsigned short*)(ws + off);
  off = alignup(off + (size_t)C_COLS * D_DIM * sizeof(unsigned short));
  const size_t need_pre = off;
  (void)in_sizes; (void)n_in; (void)out_size; (void)need_fallback;

  const bool pre = (ws_size >= need_pre);  // constant across calls -> deterministic

  rownorm_bf16_kernel<<<B_ROWS / 8, 256, 0, stream>>>(features, featbf);
  if (pre) {
    // normalize+convert weight once; GEMM then streams bf16 from L2
    rownorm_bf16_kernel<<<C_COLS / 8, 256, 0, stream>>>(weight, wbf);
    arc_gemm_kernel<true><<<dim3(NPART, B_ROWS / BM), 256, 0, stream>>>(
        featbf, wbf, nullptr, nullptr, targets, tlog, pMax, pSum);
  } else {
    w_norm_kernel<<<C_COLS / 8, 256, 0, stream>>>(weight, winv);
    arc_gemm_kernel<false><<<dim3(NPART, B_ROWS / BM), 256, 0, stream>>>(
        featbf, nullptr, weight, winv, targets, tlog, pMax, pSum);
  }
  finalize_kernel<<<1, 256, 0, stream>>>(pMax, pSum, tlog, out);
}
